// HiroLRAN_63823214019249
// MI455X (gfx1250) — matrix operand order repaired and checked by reading, not compiled, not run on hardware
//
#include <hip/hip_runtime.h>

typedef __attribute__((ext_vector_type(4)))  float          v4f;
typedef __attribute__((ext_vector_type(8)))  float          v8f;
typedef __attribute__((ext_vector_type(8)))  unsigned short v8u;
typedef __attribute__((ext_vector_type(16))) unsigned short v16u;
typedef __attribute__((ext_vector_type(16))) __bf16         v16bf;
typedef __attribute__((ext_vector_type(4)))  int            v4i;

#define LRELU_SLOPE 0.01f

// ---- CDNA5 async global<->LDS path (ASYNCcnt), guarded so both toolchains
// ---- and the host pass still compile. Falls back to plain loads/stores.
#if defined(__AMDGCN__) &&                                                    \
    __has_builtin(__builtin_amdgcn_global_load_async_to_lds_b128) &&          \
    __has_builtin(__builtin_amdgcn_global_store_async_from_lds_b128) &&       \
    __has_builtin(__builtin_amdgcn_s_wait_asynccnt)
#define USE_ASYNC_LDS 1
#else
#define USE_ASYNC_LDS 0
#endif

#if USE_ASYNC_LDS
// builtin params are int4 addrspace(1)* and int4 addrspace(3)*
typedef __attribute__((address_space(1))) v4i* gv4p;
typedef __attribute__((address_space(3))) v4i* lv4p;

__device__ __forceinline__ void async_g2l_b128(const void* g, void* l) {
    __builtin_amdgcn_global_load_async_to_lds_b128(
        (gv4p)(unsigned long long)g,
        (lv4p)(unsigned int)(unsigned long long)l,
        0, 0);
}
__device__ __forceinline__ void async_l2g_b128(void* g, const void* l) {
    __builtin_amdgcn_global_store_async_from_lds_b128(
        (gv4p)(unsigned long long)g,
        (lv4p)(unsigned int)(unsigned long long)l,
        0, 0);
}
__device__ __forceinline__ void wait_async0() { __builtin_amdgcn_s_wait_asynccnt(0); }
#endif

// ---- f32 -> bf16 (round to nearest even) ---------------------------------
__device__ __forceinline__ unsigned short f2bf(float f) {
    unsigned int u = __builtin_bit_cast(unsigned int, f);
    u += 0x7FFFu + ((u >> 16) & 1u);
    return (unsigned short)(u >> 16);
}

// ---- A-operand (16x32 bf16) per-lane load --------------------------------
// Lane holds row M = lane&15. Elements 0..7  <-> K = kbase + 8*hi + i
//                             Elements 8..15 <-> K = kbase + 16 + 8*hi + i
__device__ __forceinline__ v16bf load_a16(const unsigned short* row, int kbase, int hi) {
    v8u c0 = *(const v8u*)(row + kbase + 8 * hi);
    v8u c1 = *(const v8u*)(row + kbase + 16 + 8 * hi);
    v16u a;
#pragma unroll
    for (int i = 0; i < 8; ++i) { a[i] = c0[i]; a[i + 8] = c1[i]; }
    return __builtin_bit_cast(v16bf, a);
}

// ---- B-operand (32x16 bf16) per-lane load from transposed weight Wt[N][K] -
// Lane holds col N = lane&15 (passed in as ncol). Elements 0..7 <-> K = kbase + 8*hi + i,
// elements 8..15 <-> K = kbase + 16 + 8*hi + i
__device__ __forceinline__ v16bf load_b16(const unsigned short* __restrict__ wT,
                                          int ncol, int K, int kbase, int hi) {
    const unsigned short* p = wT + (size_t)ncol * K + kbase + 8 * hi;
    v8u c0 = *(const v8u*)p;
    v8u c1 = *(const v8u*)(p + 16);
    v16u b;
#pragma unroll
    for (int i = 0; i < 8; ++i) { b[i] = c0[i]; b[i + 8] = c1[i]; }
    return __builtin_bit_cast(v16bf, b);
}

// ---- one 16x16 output tile of act[16][K] @ Wt^T, f32 accumulate ----------
__device__ __forceinline__ v8f mlp_tile(const unsigned short* act, int K,
                                        const unsigned short* __restrict__ wT,
                                        int ncol, float bias, int lane15, int hi) {
    v8f c;
#pragma unroll
    for (int r = 0; r < 8; ++r) c[r] = bias;
    const unsigned short* arow = act + lane15 * K;
#pragma unroll 4
    for (int kk = 0; kk < K; kk += 32) {
        v16bf a = load_a16(arow, kk, hi);
        v16bf b = load_b16(wT, ncol, K, kk, hi);
        c = __builtin_amdgcn_wmma_f32_16x16x32_bf16(false, a, false, b, (short)0, c,
                                                    false, false);
    }
    return c;
}

// ===========================================================================
// Weight prep: Wt[n][kp] = bf16(W[kp][n]), zero-pad kp >= K
// ===========================================================================
__global__ void prep_kernel(const float* __restrict__ src, unsigned short* __restrict__ dst,
                            int K, int N, int Kpad) {
    int idx = blockIdx.x * 256 + threadIdx.x;
    if (idx >= N * Kpad) return;
    int n = idx / Kpad, kp = idx % Kpad;
    dst[idx] = (kp < K) ? f2bf(src[(size_t)kp * N + n]) : (unsigned short)0;
}

// ===========================================================================
// Encoder at t=0 only: 64 rows, 256->512->512->256, leaky everywhere.
// grid=4 blocks x 256 thr (8 waves). Block owns rows b = blk*16 .. +15.
// ===========================================================================
__global__ void __launch_bounds__(256)
enc_kernel(const float* __restrict__ in,
           const unsigned short* __restrict__ w1t, const float* __restrict__ b1,
           const unsigned short* __restrict__ w2t, const float* __restrict__ b2,
           const unsigned short* __restrict__ w3t, const float* __restrict__ b3,
           unsigned short* __restrict__ z0) {
    __shared__ __align__(16) unsigned short xin[16 * 256];
    __shared__ __align__(16) unsigned short h1[16 * 512];
    __shared__ __align__(16) unsigned short h2[16 * 512];
    const int tid = threadIdx.x, blk = blockIdx.x;

    // load x_t[b,0,0:256], convert to bf16
    for (int idx = tid; idx < 16 * 256; idx += 256) {
        int r = idx >> 8, k = idx & 255;
        xin[idx] = f2bf(in[(size_t)(blk * 16 + r) * 1024 * 288 + k]);
    }
    __syncthreads();

    const int lane15 = tid & 15, hi = (tid >> 4) & 1, w = tid >> 5;

    // layer 1: 256 -> 512 (32 n-tiles, 4/wave)
    for (int i = 0; i < 4; ++i) {
        int ncol = (w * 4 + i) * 16 + lane15;
        v8f c = mlp_tile(xin, 256, w1t, ncol, b1[ncol], lane15, hi);
#pragma unroll
        for (int r = 0; r < 8; ++r) {
            float v = c[r]; v = v >= 0.f ? v : LRELU_SLOPE * v;
            h1[(r + 8 * hi) * 512 + ncol] = f2bf(v);
        }
    }
    __syncthreads();

    // layer 2: 512 -> 512
    for (int i = 0; i < 4; ++i) {
        int ncol = (w * 4 + i) * 16 + lane15;
        v8f c = mlp_tile(h1, 512, w2t, ncol, b2[ncol], lane15, hi);
#pragma unroll
        for (int r = 0; r < 8; ++r) {
            float v = c[r]; v = v >= 0.f ? v : LRELU_SLOPE * v;
            h2[(r + 8 * hi) * 512 + ncol] = f2bf(v);
        }
    }
    __syncthreads();

    // layer 3: 512 -> 256 (16 n-tiles, 2/wave), leaky, write bf16 z0
    for (int i = 0; i < 2; ++i) {
        int ncol = (w * 2 + i) * 16 + lane15;
        v8f c = mlp_tile(h2, 512, w3t, ncol, b3[ncol], lane15, hi);
#pragma unroll
        for (int r = 0; r < 8; ++r) {
            float v = c[r]; v = v >= 0.f ? v : LRELU_SLOPE * v;
            z0[(blk * 16 + r + 8 * hi) * 256 + ncol] = f2bf(v);
        }
    }
}

// ===========================================================================
// Sequential scan: z[t] = z[t-1] @ A + u[t] @ B.  One block, 1024 thr = 32
// waves.  State z (64x256 bf16 = 32KB) lives in LDS; Bu computed on the fly
// (u padded K 16->32 with zeros).  z1[t][b][c] bf16 is streamed out with
// GLOBAL_STORE_ASYNC_FROM_LDS_B128 overlapped with the next step's WMMAs.
// Wave w owns m-tile mt = w>>3 and n-tiles 2*(w&7), 2*(w&7)+1.
// Four independent WMMA accumulator chains hide the WMMA RAW latency.
// ===========================================================================
__global__ void __launch_bounds__(1024)
scan_kernel(const float* __restrict__ in,
            const unsigned short* __restrict__ Awt,   // [256][256]
            const unsigned short* __restrict__ Bwt,   // [256][32] (K padded)
            const unsigned short* __restrict__ z0,
            unsigned short* __restrict__ z1) {
    __shared__ __align__(16) unsigned short z[64 * 256];
    const int tid = threadIdx.x;

#if USE_ASYNC_LDS
    {   // async preload of the initial state (2 x b128 per thread)
        char* lbase = (char*)z;
        const char* gbase = (const char*)z0;
#pragma unroll
        for (int j = 0; j < 2; ++j) {
            int chunk = tid * 2 + j;
            async_g2l_b128(gbase + chunk * 16, lbase + chunk * 16);
        }
        wait_async0();
    }
#else
    for (int idx = tid; idx < 64 * 256; idx += 1024) z[idx] = z0[idx];
#endif
    __syncthreads();

    const int lane15 = tid & 15, hi = (tid >> 4) & 1, w = tid >> 5;
    const int mt = w >> 3;
    const int n0 = ((w & 7) * 2) * 16 + lane15;
    const int n1 = n0 + 16;
    const int brow = mt * 16 + lane15;                    // A-operand row for this lane
    const unsigned short* arow = &z[brow * 256];
    const float* ubase = in + (size_t)brow * 1024 * 288 + 272 + 8 * hi;

    for (int t = 0; t < 1024; ++t) {
        // --- Bu: A-operand = u row (8 real values + 16..31 zero pad) ---
        v4f ua = *(const v4f*)(ubase + (size_t)t * 288);
        v4f ub = *(const v4f*)(ubase + (size_t)t * 288 + 4);
        v16u au;
#pragma unroll
        for (int i = 0; i < 4; ++i) {
            au[i] = f2bf(ua[i]); au[4 + i] = f2bf(ub[i]);
            au[8 + i] = 0;       au[12 + i] = 0;
        }
        v16bf abf = __builtin_bit_cast(v16bf, au);

        v8f c0a, c0b, c1a, c1b;
#pragma unroll
        for (int r = 0; r < 8; ++r) { c0a[r] = 0.f; c0b[r] = 0.f; c1a[r] = 0.f; c1b[r] = 0.f; }
        c0a = __builtin_amdgcn_wmma_f32_16x16x32_bf16(false, abf, false,
                  load_b16(Bwt, n0, 32, 0, hi), (short)0, c0a, false, false);
        c1a = __builtin_amdgcn_wmma_f32_16x16x32_bf16(false, abf, false,
                  load_b16(Bwt, n1, 32, 0, hi), (short)0, c1a, false, false);

        // --- z @ A : K = 256, 4 independent accumulator chains ---
#pragma unroll
        for (int kk = 0; kk < 256; kk += 64) {
            v16bf a0 = load_a16(arow, kk, hi);
            v16bf a1 = load_a16(arow, kk + 32, hi);
            c0a = __builtin_amdgcn_wmma_f32_16x16x32_bf16(false, a0, false,
                      load_b16(Awt, n0, 256, kk, hi), (short)0, c0a, false, false);
            c1a = __builtin_amdgcn_wmma_f32_16x16x32_bf16(false, a0, false,
                      load_b16(Awt, n1, 256, kk, hi), (short)0, c1a, false, false);
            c0b = __builtin_amdgcn_wmma_f32_16x16x32_bf16(false, a1, false,
                      load_b16(Awt, n0, 256, kk + 32, hi), (short)0, c0b, false, false);
            c1b = __builtin_amdgcn_wmma_f32_16x16x32_bf16(false, a1, false,
                      load_b16(Awt, n1, 256, kk + 32, hi), (short)0, c1b, false, false);
        }
        v8f c0 = c0a + c0b;
        v8f c1 = c1a + c1b;

#if USE_ASYNC_LDS
        wait_async0();       // step t-1's async LDS->global stores done reading z
#endif
        __syncthreads();     // everyone done reading old z
#pragma unroll
        for (int r = 0; r < 8; ++r) {
            int br = mt * 16 + r + 8 * hi;
            unsigned short h0 = f2bf(c0[r]);
            unsigned short h1v = f2bf(c1[r]);
            z[br * 256 + n0] = h0;
            z[br * 256 + n1] = h1v;
#if !USE_ASYNC_LDS
            z1[((size_t)t * 64 + br) * 256 + n0] = h0;
            z1[((size_t)t * 64 + br) * 256 + n1] = h1v;
#endif
        }
        __syncthreads();     // new z visible to all waves (and to async stores)
#if USE_ASYNC_LDS
        {   // stream z[t] (32KB contiguous) to global, overlapped with step t+1
            char* gbase = (char*)z1 + (size_t)t * 64 * 256 * 2;
            const char* lbase = (const char*)z;
#pragma unroll
            for (int j = 0; j < 2; ++j) {
                int chunk = tid * 2 + j;
                async_l2g_b128(gbase + chunk * 16, lbase + chunk * 16);
            }
        }
#endif
    }
}

// ===========================================================================
// Decoder: 65536 rows, 256->512->512->256, leaky on first two layers.
// grid = 4096 blocks (blk = t*4 + bg), block owns rows t*64 + bg*16 .. +15.
// ===========================================================================
__global__ void __launch_bounds__(256)
dec_kernel(const unsigned short* __restrict__ z1,
           const unsigned short* __restrict__ w1t, const float* __restrict__ b1,
           const unsigned short* __restrict__ w2t, const float* __restrict__ b2,
           const unsigned short* __restrict__ w3t, const float* __restrict__ b3,
           float* __restrict__ out) {
    __shared__ __align__(16) unsigned short zin[16 * 256];
    __shared__ __align__(16) unsigned short h1[16 * 512];
    __shared__ __align__(16) unsigned short h2[16 * 512];
    const int tid = threadIdx.x, blk = blockIdx.x;
    const int t = blk >> 2, bg = blk & 3;

#if USE_ASYNC_LDS
    {   // async copy of the contiguous 8KB z1 tile into LDS (2 x b128/thread)
        const char* gbase = (const char*)z1 + (size_t)blk * 16 * 256 * 2;
        char* lbase = (char*)zin;
#pragma unroll
        for (int j = 0; j < 2; ++j) {
            int chunk = tid * 2 + j;
            async_g2l_b128(gbase + chunk * 16, lbase + chunk * 16);
        }
        wait_async0();
    }
#else
    {
        const unsigned int* src = (const unsigned int*)(z1 + (size_t)blk * 16 * 256);
        unsigned int* dst = (unsigned int*)zin;
        for (int idx = tid; idx < 2048; idx += 256) dst[idx] = src[idx];
    }
#endif
    __syncthreads();

    const int lane15 = tid & 15, hi = (tid >> 4) & 1, w = tid >> 5;

    for (int i = 0; i < 4; ++i) {
        int ncol = (w * 4 + i) * 16 + lane15;
        v8f c = mlp_tile(zin, 256, w1t, ncol, b1[ncol], lane15, hi);
#pragma unroll
        for (int r = 0; r < 8; ++r) {
            float v = c[r]; v = v >= 0.f ? v : LRELU_SLOPE * v;
            h1[(r + 8 * hi) * 512 + ncol] = f2bf(v);
        }
    }
    __syncthreads();

    for (int i = 0; i < 4; ++i) {
        int ncol = (w * 4 + i) * 16 + lane15;
        v8f c = mlp_tile(h1, 512, w2t, ncol, b2[ncol], lane15, hi);
#pragma unroll
        for (int r = 0; r < 8; ++r) {
            float v = c[r]; v = v >= 0.f ? v : LRELU_SLOPE * v;
            h2[(r + 8 * hi) * 512 + ncol] = f2bf(v);
        }
    }
    __syncthreads();

    // layer 3: no activation, f32 output to out[b][t][c]
    for (int i = 0; i < 2; ++i) {
        int ncol = (w * 2 + i) * 16 + lane15;
        v8f c = mlp_tile(h2, 512, w3t, ncol, b3[ncol], lane15, hi);
#pragma unroll
        for (int r = 0; r < 8; ++r) {
            int bIdx = bg * 16 + r + 8 * hi;
            out[((size_t)bIdx * 1024 + t) * 256 + ncol] = c[r];
        }
    }
}

// ===========================================================================
extern "C" void kernel_launch(void* const* d_in, const int* in_sizes, int n_in,
                              void* d_out, int out_size, void* d_ws, size_t ws_size,
                              hipStream_t stream) {
    (void)in_sizes; (void)n_in; (void)out_size; (void)ws_size;
    const float* in    = (const float*)d_in[0];
    const float* ew1   = (const float*)d_in[1];
    const float* eb1   = (const float*)d_in[2];
    const float* ew2   = (const float*)d_in[3];
    const float* eb2   = (const float*)d_in[4];
    const float* ew3   = (const float*)d_in[5];
    const float* eb3   = (const float*)d_in[6];
    const float* A_w   = (const float*)d_in[7];
    const float* B_w   = (const float*)d_in[8];
    const float* dw1   = (const float*)d_in[9];
    const float* db1   = (const float*)d_in[10];
    const float* dw2   = (const float*)d_in[11];
    const float* db2   = (const float*)d_in[12];
    const float* dw3   = (const float*)d_in[13];
    const float* db3   = (const float*)d_in[14];

    char* ws = (char*)d_ws;
    size_t off = 0;
    auto alloc = [&](size_t bytes) -> void* {
        void* p = ws + off;
        off += (bytes + 255) & ~(size_t)255;
        return p;
    };
    unsigned short* enc_w1t = (unsigned short*)alloc((size_t)512 * 256 * 2);
    unsigned short* enc_w2t = (unsigned short*)alloc((size_t)512 * 512 * 2);
    unsigned short* enc_w3t = (unsigned short*)alloc((size_t)256 * 512 * 2);
    unsigned short* A_wt    = (unsigned short*)alloc((size_t)256 * 256 * 2);
    unsigned short* B_wt    = (unsigned short*)alloc((size_t)256 * 32 * 2);
    unsigned short* dec_w1t = (unsigned short*)alloc((size_t)512 * 256 * 2);
    unsigned short* dec_w2t = (unsigned short*)alloc((size_t)512 * 512 * 2);
    unsigned short* dec_w3t = (unsigned short*)alloc((size_t)256 * 512 * 2);
    unsigned short* z0      = (unsigned short*)alloc((size_t)64 * 256 * 2);
    unsigned short* z1      = (unsigned short*)alloc((size_t)1024 * 64 * 256 * 2);

    auto prep = [&](const float* s, unsigned short* d, int K, int N, int Kpad) {
        int tot = N * Kpad;
        prep_kernel<<<(tot + 255) / 256, 256, 0, stream>>>(s, d, K, N, Kpad);
    };
    prep(ew1, enc_w1t, 256, 512, 256);
    prep(ew2, enc_w2t, 512, 512, 512);
    prep(ew3, enc_w3t, 512, 256, 512);
    prep(A_w, A_wt,    256, 256, 256);
    prep(B_w, B_wt,     16, 256,  32);
    prep(dw1, dec_w1t, 256, 512, 256);
    prep(dw2, dec_w2t, 512, 512, 512);
    prep(dw3, dec_w3t, 512, 256, 512);

    enc_kernel<<<4, 256, 0, stream>>>(in, enc_w1t, eb1, enc_w2t, eb2, enc_w3t, eb3, z0);
    scan_kernel<<<1, 1024, 0, stream>>>(in, A_wt, B_wt, z0, z1);
    dec_kernel<<<4096, 256, 0, stream>>>(z1, dec_w1t, db1, dec_w2t, db2, dec_w3t, db3,
                                         (float*)d_out);
}
